// MultiHeadAttention_36447092474105
// MI455X (gfx1250) — matrix operand order repaired and checked by reading, not compiled, not run on hardware
//
#include <hip/hip_runtime.h>
#include <hip/hip_bf16.h>

typedef __bf16 bf16_t;
typedef __attribute__((ext_vector_type(16))) __bf16 v16bf;
typedef __attribute__((ext_vector_type(8)))  __bf16 v8bf;
typedef __attribute__((ext_vector_type(8)))  float  v8f;

#define WMMA_BF16(A, B, C) \
    __builtin_amdgcn_wmma_f32_16x16x32_bf16(false, (A), false, (B), (short)0, (C), false, false)

// ---------------------------------------------------------------------------
// CDNA5 async global->LDS copy (ASYNCcnt-tracked), inline asm per ISA §10/§15.18.
// Copies 16 bytes per active lane: LDS[lds_byte_addr] = MEM[gaddr].
// ---------------------------------------------------------------------------
__device__ __forceinline__ void async_copy16(unsigned int lds_byte_addr, unsigned long long gaddr) {
    asm volatile("global_load_async_to_lds_b128 %0, %1, off"
                 :: "v"(lds_byte_addr), "v"(gaddr) : "memory");
}
__device__ __forceinline__ void wait_async_le16() {
    asm volatile("s_wait_asynccnt 0x10" ::: "memory");
}
__device__ __forceinline__ void wait_async_0() {
    asm volatile("s_wait_asynccnt 0x0" ::: "memory");
}

// ---------------------------------------------------------------------------
// Fragment loaders (wave32, ISA §7.12.2 layouts)
// A 16x32 bf16: lane L -> row m=L&15; elems 0..7 = K=half*8.., elems 8..15 = K+16
// ---------------------------------------------------------------------------
__device__ __forceinline__ v16bf load_a_frag(const bf16_t* base, int ld) {
    int lane = threadIdx.x & 31;
    int m    = lane & 15;
    int half = lane >> 4;
    const bf16_t* p0 = base + (size_t)m * ld + half * 8;
    v8bf lo = *(const v8bf*)p0;
    v8bf hi = *(const v8bf*)(p0 + 16);
    v16bf out;
#pragma unroll
    for (int i = 0; i < 8; ++i) { out[i] = lo[i]; out[i + 8] = hi[i]; }
    return out;
}

// B 32x16 bf16 from transposed storage Bt[n*ld + k]:
// lane L -> col n=L&15; elems 0..7 = K=half*8.., elems 8..15 = K+16
__device__ __forceinline__ v16bf load_b_frag(const bf16_t* baseT, int ld) {
    int lane = threadIdx.x & 31;
    int n    = lane & 15;
    int half = lane >> 4;
    const bf16_t* p0 = baseT + (size_t)n * ld + half * 8;
    v8bf lo = *(const v8bf*)p0;
    v8bf hi = *(const v8bf*)(p0 + 16);
    v16bf out;
#pragma unroll
    for (int i = 0; i < 8; ++i) { out[i] = lo[i]; out[i + 8] = hi[i]; }
    return out;
}

// ---------------------------------------------------------------------------
// Stage 1: f32 -> bf16 conversions
// ---------------------------------------------------------------------------
__global__ void mha_f32_to_bf16(const float* __restrict__ in, bf16_t* __restrict__ out, int n) {
    int i = blockIdx.x * blockDim.x + threadIdx.x;
    if (i < n) out[i] = (bf16_t)in[i];
}

// W [K,N] f32 -> Wt [N,K] bf16
__global__ void mha_transpose_bf16(const float* __restrict__ W, bf16_t* __restrict__ Wt,
                                   int K, int N) {
    int i = blockIdx.x * blockDim.x + threadIdx.x;
    if (i < K * N) {
        int k = i / N, n = i - k * N;
        Wt[(size_t)n * K + k] = (bf16_t)W[i];
    }
}

// ---------------------------------------------------------------------------
// Stage 2/5: GEMM  C[M,N] = A[M,K] (bf16 row-major) x Bt[N,K] (bf16)
// wave tile: 32x64 (2x4 wmma tiles), block = 4 waves along N.
// Output type is a template parameter -> branch-free epilogue.
// ---------------------------------------------------------------------------
template <bool F32OUT>
__launch_bounds__(128)
__global__ void mha_gemm_bf16(const bf16_t* __restrict__ A, const bf16_t* __restrict__ Bt,
                              void* __restrict__ Cout, int M, int N, int K) {
    int lane = threadIdx.x & 31;
    int wave = threadIdx.x >> 5;
    int m0 = blockIdx.y * 32;
    int n0 = (blockIdx.x * 4 + wave) * 64;

    v8f acc[2][4] = {};
    for (int k0 = 0; k0 < K; k0 += 32) {
        v16bf af[2];
#pragma unroll
        for (int mt = 0; mt < 2; ++mt)
            af[mt] = load_a_frag(A + (size_t)(m0 + mt * 16) * K + k0, K);
        v16bf bfrag[4];
#pragma unroll
        for (int nt = 0; nt < 4; ++nt)
            bfrag[nt] = load_b_frag(Bt + (size_t)(n0 + nt * 16) * K + k0, K);
#pragma unroll
        for (int mt = 0; mt < 2; ++mt)
#pragma unroll
            for (int nt = 0; nt < 4; ++nt)
                acc[mt][nt] = WMMA_BF16(af[mt], bfrag[nt], acc[mt][nt]);
    }

    int half = lane >> 4, cn = lane & 15;
#pragma unroll
    for (int mt = 0; mt < 2; ++mt)
#pragma unroll
        for (int r = 0; r < 8; ++r) {
            size_t rowoff = (size_t)(m0 + mt * 16 + r + 8 * half) * N + n0 + cn;
            if constexpr (F32OUT) {
                float* Cf = (float*)Cout;
#pragma unroll
                for (int nt = 0; nt < 4; ++nt)
                    Cf[rowoff + nt * 16] = acc[mt][nt][r];
            } else {
                bf16_t* Cb = (bf16_t*)Cout;
#pragma unroll
                for (int nt = 0; nt < 4; ++nt)
                    Cb[rowoff + nt * 16] = (bf16_t)acc[mt][nt][r];
            }
        }
}

// ---------------------------------------------------------------------------
// Stage 3: RoPE + head reshape.  qkv bf16 [B,T,3C] ->
//   qh [B,H,T,D] (scaled by 1/sqrt(D)), kh [B,H,T,D], vT [B,H,D,T]
// ---------------------------------------------------------------------------
__global__ void mha_rope(const bf16_t* __restrict__ qkv,
                         bf16_t* __restrict__ qh, bf16_t* __restrict__ kh,
                         bf16_t* __restrict__ vT) {
    const int T = 2048, H = 16, D = 64, C = 1024;
    int idx = blockIdx.x * blockDim.x + threadIdx.x;   // B*T*H*D = 4M
    int d = idx & 63;
    int h = (idx >> 6) & 15;
    int t = (idx >> 10) & 2047;
    int b = idx >> 21;

    size_t rowb = ((size_t)(b * T + t)) * (3 * C);
    int col = h * D + d;
    float qv = (float)qkv[rowb + col];
    float kv = (float)qkv[rowb + C + col];
    float vv = (float)qkv[rowb + 2 * C + col];

    int dh = d & 31;
    float inv_freq = __expf(-(float)(2 * dh) * (9.210340371976184f / 64.0f)); // 10000^(-2i/64)
    float ang = (float)t * inv_freq;
    float s, c;
    __sincosf(ang, &s, &c);

    int dpart = (d < 32) ? d + 32 : d - 32;
    float sgn = (d < 32) ? -1.0f : 1.0f;
    float qp = (float)qkv[rowb + h * D + dpart];
    float kp = (float)qkv[rowb + C + h * D + dpart];

    float qo = (qv * c + sgn * qp * s) * 0.125f;       // fold 1/sqrt(64)
    float ko = (kv * c + sgn * kp * s);

    size_t bh = (size_t)(b * H + h);
    qh[(bh * T + t) * D + d] = (bf16_t)qo;
    kh[(bh * T + t) * D + d] = (bf16_t)ko;
    vT[(bh * D + d) * T + t] = (bf16_t)vv;
}

// ---------------------------------------------------------------------------
// Stage 4: causal flash attention.  One wave per (b,h,16-query tile).
// K/V tiles (32 keys) stream into LDS via global_load_async_to_lds_b128,
// double-buffered and pipelined on ASYNCcnt.  S = Q K^T (4 wmma / 32 keys),
// online softmax in f32, P staged via LDS into A-fragment, O += P V (4 wmma).
// ---------------------------------------------------------------------------
__launch_bounds__(128)
__global__ void mha_flash_attn(const bf16_t* __restrict__ qh, const bf16_t* __restrict__ kh,
                               const bf16_t* __restrict__ vT, bf16_t* __restrict__ attn) {
    const int T = 2048, H = 16, D = 64;
    // per-wave double buffers: K tile 32x64 bf16, V tile 64x32 bf16, P tile 16x32
    __shared__ __align__(16) bf16_t kbuf[4][2][32 * 64];
    __shared__ __align__(16) bf16_t vbuf[4][2][64 * 32];
    __shared__ __align__(16) bf16_t plds[4][16 * 32];

    int lane = threadIdx.x & 31;
    int wave = threadIdx.x >> 5;
    int wid  = blockIdx.x * 4 + wave;
    int qt = wid & 127;
    int h  = (wid >> 7) & 15;
    int b  = wid >> 11;

    size_t bh = (size_t)(b * H + h);
    const bf16_t* qb = qh + bh * T * D;
    const bf16_t* kb = kh + bh * T * D;
    const bf16_t* vb = vT + bh * D * T;

    int q0 = qt * 16;
    v16bf qa[2];
    qa[0] = load_a_frag(qb + (size_t)q0 * D + 0,  D);
    qa[1] = load_a_frag(qb + (size_t)q0 * D + 32, D);

    int half = lane >> 4, cn = lane & 15;
    float mrow[8], lrow[8];
    v8f o[4] = {};
#pragma unroll
    for (int r = 0; r < 8; ++r) { mrow[r] = -1e30f; lrow[r] = 0.0f; }

    bf16_t* pw = plds[wave];
    int qmax = q0 + 15;

    // ---- async K/V tile issue: 16 x b128 per tile (4KB K + 4KB V) ----
    auto issue_tile = [&](int j, int buf) {
        unsigned int kdst = (unsigned int)(size_t)&kbuf[wave][buf][0];
        unsigned int vdst = (unsigned int)(size_t)&vbuf[wave][buf][0];
        const char* kbase = (const char*)kb;
        const char* vbase = (const char*)vb;
#pragma unroll
        for (int qq = 0; qq < 8; ++qq) {
            int chunk = qq * 32 + lane;                        // 0..255, 16B each
            // K: rows j..j+31 (128B/row) -> kbuf row-major 32x64
            const char* gk = kbase + (size_t)(j + (chunk >> 3)) * 128 + (size_t)(chunk & 7) * 16;
            async_copy16(kdst + (unsigned int)chunk * 16, (unsigned long long)(size_t)gk);
            // V: vT rows d=0..63, 64B segment at key j -> vbuf row-major 64x32
            const char* gv = vbase + (size_t)(chunk >> 2) * (T * 2) + (size_t)j * 2
                                   + (size_t)(chunk & 3) * 16;
            async_copy16(vdst + (unsigned int)chunk * 16, (unsigned long long)(size_t)gv);
        }
    };

    int cur = 0;
    issue_tile(0, 0);

    for (int j = 0; j <= qmax; j += 32) {
        if (j + 32 <= qmax) {
            issue_tile(j + 32, cur ^ 1);
            wait_async_le16();      // previous 16 (current tile) complete; prefetch in flight
        } else {
            wait_async_0();
        }
        const bf16_t* kc = kbuf[wave][cur];
        const bf16_t* vc = vbuf[wave][cur];

        // ---- S = q . k^T for 32 keys (two 16-key tiles), fragments from LDS ----
        v8f s[2];
#pragma unroll
        for (int st = 0; st < 2; ++st) {
            v8f sa = {};
            v16bf kf0 = load_b_frag(kc + (st * 16) * 64 + 0,  64);
            v16bf kf1 = load_b_frag(kc + (st * 16) * 64 + 32, 64);
            sa = WMMA_BF16(qa[0], kf0, sa);
            sa = WMMA_BF16(qa[1], kf1, sa);
            s[st] = sa;
        }

        // ---- causal mask + online softmax (rows = r + 8*half) ----
#pragma unroll
        for (int r = 0; r < 8; ++r) {
            int qrow = q0 + r + 8 * half;
            float v0 = (j + cn      <= qrow) ? s[0][r] : -1e30f;
            float v1 = (j + 16 + cn <= qrow) ? s[1][r] : -1e30f;

            float tm = fmaxf(v0, v1);
            tm = fmaxf(tm, __shfl_xor(tm, 1, 16));
            tm = fmaxf(tm, __shfl_xor(tm, 2, 16));
            tm = fmaxf(tm, __shfl_xor(tm, 4, 16));
            tm = fmaxf(tm, __shfl_xor(tm, 8, 16));

            float mnew  = fmaxf(mrow[r], tm);
            float alpha = __expf(mrow[r] - mnew);
            mrow[r] = mnew;

            float p0 = __expf(v0 - mnew);
            float p1 = __expf(v1 - mnew);
            float ps = p0 + p1;
            ps += __shfl_xor(ps, 1, 16);
            ps += __shfl_xor(ps, 2, 16);
            ps += __shfl_xor(ps, 4, 16);
            ps += __shfl_xor(ps, 8, 16);
            lrow[r] = lrow[r] * alpha + ps;

            int row = r + 8 * half;
            pw[row * 32 + cn]      = (bf16_t)p0;
            pw[row * 32 + 16 + cn] = (bf16_t)p1;

#pragma unroll
            for (int nt = 0; nt < 4; ++nt) o[nt][r] *= alpha;
        }

        // ---- reload P as A-fragment (16x32) from LDS ----
        v16bf pa;
        {
            const bf16_t* p0 = pw + (lane & 15) * 32 + half * 8;
            v8bf lo = *(const v8bf*)p0;
            v8bf hi = *(const v8bf*)(p0 + 16);
#pragma unroll
            for (int i = 0; i < 8; ++i) { pa[i] = lo[i]; pa[i + 8] = hi[i]; }
        }

        // ---- O += P . V from LDS V tile (rows contiguous in key) ----
#pragma unroll
        for (int nt = 0; nt < 4; ++nt) {
            v16bf vf = load_b_frag(vc + (nt * 16) * 32, 32);
            o[nt] = WMMA_BF16(pa, vf, o[nt]);
        }
        cur ^= 1;
    }

    // ---- normalize + store bf16 into [B,T,H*D] ----
#pragma unroll
    for (int r = 0; r < 8; ++r) {
        float inv = 1.0f / lrow[r];
        int row = q0 + r + 8 * half;
#pragma unroll
        for (int nt = 0; nt < 4; ++nt) {
            int col = h * D + nt * 16 + cn;
            attn[((size_t)(b * T + row)) * (H * D) + col] = (bf16_t)(o[nt][r] * inv);
        }
    }
}

// ---------------------------------------------------------------------------
// Orchestration
// ---------------------------------------------------------------------------
extern "C" void kernel_launch(void* const* d_in, const int* in_sizes, int n_in,
                              void* d_out, int out_size, void* d_ws, size_t ws_size,
                              hipStream_t stream) {
    const int B = 2, T = 2048, H = 16, D = 64, C = H * D;   // 1024
    const int M = B * T;                                    // 4096

    const float* x      = (const float*)d_in[0];   // [B,T,C]
    const float* W_qkv  = (const float*)d_in[1];   // [C,3C]
    const float* W_proj = (const float*)d_in[2];   // [C,C]
    float* out = (float*)d_out;                    // [B,T,C] f32

    char* ws = (char*)d_ws;
    bf16_t* xb      = (bf16_t*)(ws + (size_t)0);            //  8 MiB: [M,C]
    bf16_t* wqkvT   = (bf16_t*)(ws + ((size_t)8  << 20));   //  6 MiB: [3C,C]
    bf16_t* wprojT  = (bf16_t*)(ws + ((size_t)14 << 20));   //  2 MiB: [C,C]
    bf16_t* qkv     = (bf16_t*)(ws + ((size_t)16 << 20));   // 24 MiB: [M,3C]
    bf16_t* qhb     = (bf16_t*)(ws + ((size_t)40 << 20));   //  8 MiB: [B,H,T,D]
    bf16_t* khb     = (bf16_t*)(ws + ((size_t)48 << 20));   //  8 MiB: [B,H,T,D]
    bf16_t* vTb     = (bf16_t*)(ws + ((size_t)56 << 20));   //  8 MiB: [B,H,D,T]
    bf16_t* attnb   = (bf16_t*)(ws + ((size_t)64 << 20));   //  8 MiB: [M,C]

    // Stage 1: conversions
    {
        int n = M * C;
        mha_f32_to_bf16<<<(n + 255) / 256, 256, 0, stream>>>(x, xb, n);
        mha_transpose_bf16<<<(C * 3 * C + 255) / 256, 256, 0, stream>>>(W_qkv, wqkvT, C, 3 * C);
        mha_transpose_bf16<<<(C * C + 255) / 256, 256, 0, stream>>>(W_proj, wprojT, C, C);
    }

    // Stage 2: qkv = x @ W_qkv   (M=4096, N=3072, K=1024), bf16 out
    mha_gemm_bf16<false><<<dim3((3 * C) / 256, M / 32), 128, 0, stream>>>(
        xb, wqkvT, qkv, M, 3 * C, C);

    // Stage 3: RoPE + head split (+ v transpose)
    mha_rope<<<(B * T * H * D) / 256, 256, 0, stream>>>(qkv, qhb, khb, vTb);

    // Stage 4: causal flash attention; 4096 waves, 4 waves/block
    mha_flash_attn<<<(B * H * (T / 16)) / 4, 128, 0, stream>>>(qhb, khb, vTb, attnb);

    // Stage 5: out = attn @ W_proj  (M=4096, N=1024, K=1024), f32 out
    mha_gemm_bf16<true><<<dim3(C / 256, M / 32), 128, 0, stream>>>(
        attnb, wprojT, out, M, C, C);
}
